// MobileTernaryNet_16363825397995
// MI455X (gfx1250) — compile-verified, matrix operand order checked by reading, not run on hardware
//
#include <hip/hip_runtime.h>

typedef __attribute__((ext_vector_type(16))) _Float16 v16h;
typedef __attribute__((ext_vector_type(8)))  _Float16 v8h;
typedef __attribute__((ext_vector_type(8)))  float    v8f;

#define BN_EPS 1e-5f

// ---------------------------------------------------------------------------
// Ternarize a weight tensor (TWN): delta = 0.7*mean|w|, alpha = mean|w| over
// kept entries.  Conv weights are emitted as f16 {-1,0,+1} in WMMA A-fragment
// layout [kstep][mtile][lane][16] with TAP-MAJOR K ordering:
//   k' = (r*3+s)*CIN + c   (matches NHWC activation fragments)
// FC weights are emitted flat.  alpha is written for later folding.
// ---------------------------------------------------------------------------
__global__ void ternarize_kernel(const float* __restrict__ w, int n,
                                 int cout, int cin, int ksteps,
                                 _Float16* __restrict__ frag_out,
                                 _Float16* __restrict__ flat_out,
                                 float* __restrict__ alpha_out)
{
    __shared__ float red0[256];
    __shared__ float red1[256];
    const int t = threadIdx.x;

    // pass 1: mean |w|
    float s = 0.f;
    for (int i = t; i < n; i += 256) s += fabsf(w[i]);
    red0[t] = s;
    __syncthreads();
    for (int o = 128; o > 0; o >>= 1) {
        if (t < o) red0[t] += red0[t + o];
        __syncthreads();
    }
    const float delta = 0.7f * red0[0] / (float)n;
    __syncthreads();

    // pass 2: alpha = sum(|w| kept) / max(count,1)
    float sk = 0.f, ck = 0.f;
    for (int i = t; i < n; i += 256) {
        float a = fabsf(w[i]);
        if (a > delta) { sk += a; ck += 1.f; }
    }
    red0[t] = sk; red1[t] = ck;
    __syncthreads();
    for (int o = 128; o > 0; o >>= 1) {
        if (t < o) { red0[t] += red0[t + o]; red1[t] += red1[t + o]; }
        __syncthreads();
    }
    const float alpha = red0[0] / fmaxf(red1[0], 1.f);
    if (t == 0) alpha_out[0] = alpha;

    // pass 3: emit {-1,0,+1} f16
    if (frag_out) {
        const int kdim   = cin * 9;
        const int mtiles = cout >> 4;
        const int total  = ksteps * mtiles * 32 * 16;
        for (int e = t; e < total; e += 256) {
            int h    = e & 15;
            int lane = (e >> 4) & 31;
            int mt   = (e >> 9) % mtiles;
            int ks   = e / (512 * mtiles);
            int m    = mt * 16 + (lane & 15);
            int g    = lane >> 4;
            int k    = ks * 32 + (h & 7) + 8 * g + 16 * (h >> 3);  // tap-major k'
            float q  = 0.f;
            if (k < kdim) {
                int tap = k / cin;          // r*3+s
                int c   = k - tap * cin;    // input channel
                float wv = w[(m * cin + c) * 9 + tap];   // OIHW: [m][c][r][s]
                if (fabsf(wv) > delta) q = (wv > 0.f) ? 1.f : -1.f;
            }
            frag_out[e] = (_Float16)q;
        }
    }
    if (flat_out) {
        for (int i = t; i < n; i += 256) {
            float wv = w[i];
            float q  = (fabsf(wv) > delta) ? ((wv > 0.f) ? 1.f : -1.f) : 0.f;
            flat_out[i] = (_Float16)q;
        }
    }
}

// ---------------------------------------------------------------------------
// Fold ternary alpha + conv bias + BN into per-channel scale/bias:
//   y = relu(conv_q * (alpha*inv) + (b*inv + beta - m*inv)),  inv = g*rsqrt(v+eps)
// ---------------------------------------------------------------------------
__global__ void bnfold_kernel(const float* __restrict__ alpha,
                              const float* __restrict__ bconv,
                              const float* __restrict__ gg,
                              const float* __restrict__ beta,
                              const float* __restrict__ mm,
                              const float* __restrict__ vv,
                              float* __restrict__ sc, float* __restrict__ bi, int c)
{
    int i = blockIdx.x * blockDim.x + threadIdx.x;
    if (i < c) {
        float inv = gg[i] * rsqrtf(vv[i] + BN_EPS);
        sc[i] = alpha[0] * inv;
        bi[i] = bconv[i] * inv + beta[i] - mm[i] * inv;
    }
}

// ---------------------------------------------------------------------------
// Implicit-GEMM 3x3 conv (pad 1) via v_wmma_f32_16x16x32_f16.
//   D[Cout, Npix] = Wq[Cout, 9*CIN] x im2col(X)[9*CIN, Npix], tap-major K.
// Block = 256 threads = 8 waves; each wave owns 16 consecutive flattened
// output pixels of one image and MT 16-row Cout tiles.
//
// Fast path (CIN % 32 == 0, f16 NHWC input): each k-step is one filter tap
// (or half of one for CIN=64); a lane's B fragment = two contiguous 16B
// vector loads + one bounds check.  Slow path (conv1, f32 NCHW harness
// input): per-element gather.  Fused alpha/bias/BN/ReLU epilogue emits
// packed v8h NHWC stores.
// ---------------------------------------------------------------------------
template <typename TIN, int CIN, int COUT, int MT, int STRIDE,
          int H, int W, int OH, int OW>
__global__ __launch_bounds__(256) void conv_wmma_kernel(
    const TIN* __restrict__ in,
    const _Float16* __restrict__ Afrag,
    const float* __restrict__ sc,
    const float* __restrict__ bi,
    _Float16* __restrict__ out)
{
    constexpr int KDIM   = CIN * 9;
    constexpr int KSTEPS = (KDIM + 31) / 32;
    constexpr int MTILES = COUT / 16;
    constexpr int PPI    = (OH * OW) / 16;   // 16-pixel tiles per image

    const int lane = threadIdx.x & 31;
    const int wave = threadIdx.x >> 5;

    const int nt = blockIdx.x * 8 + wave;    // global pixel-tile id
    const int b  = nt / PPI;
    const int p0 = (nt - b * PPI) * 16;
    const int mbase_t = blockIdx.y * MT;     // base 16-row M tile

    const int nl = lane & 15;
    const int g  = lane >> 4;
    const int p  = p0 + nl;                  // this lane's output pixel
    const int oh = p / OW;
    const int ow = p - oh * OW;

    v8f acc[MT];
#pragma unroll
    for (int i = 0; i < MT; ++i) acc[i] = (v8f){0.f,0.f,0.f,0.f,0.f,0.f,0.f,0.f};

#pragma unroll
    for (int ks = 0; ks < KSTEPS; ++ks) {
        v16h bf;
        if constexpr ((CIN % 32) == 0) {
            // ---- fast path: NHWC f16, one tap per 32-wide k-step ----
            constexpr int SUBS = CIN / 32;
            const int tap = ks / SUBS;             // constants after unroll
            const int cb  = (ks - tap * SUBS) * 32;
            const int r   = tap / 3;
            const int s   = tap - r * 3;
            const int ih  = oh * STRIDE - 1 + r;
            const int iw  = ow * STRIDE - 1 + s;
            const bool ok = (ih >= 0) & (ih < H) & (iw >= 0) & (iw < W);
            const TIN* pp = in + (((long)b * H + ih) * W + iw) * CIN + cb + 8 * g;
            const v8h z = (v8h){0,0,0,0,0,0,0,0};
            union { v16h v; v8h c[2]; } u;
            u.c[0] = ok ? *(const v8h*)(pp)      : z;   // channels cb+8g   .. +7
            u.c[1] = ok ? *(const v8h*)(pp + 16) : z;   // channels cb+16+8g.. +7
            bf = u.v;
        } else {
            // ---- slow path (conv1): NCHW f32 gather, tap-major k' ----
            const TIN* inb = in + (long)b * CIN * H * W;
#pragma unroll
            for (int h = 0; h < 16; ++h) {
                int k   = ks * 32 + (h & 7) + 8 * g + 16 * (h >> 3);
                int tap = k / CIN;
                int c   = k - tap * CIN;
                int r   = tap / 3;
                int s   = tap - r * 3;
                int ih  = oh * STRIDE - 1 + r;
                int iw  = ow * STRIDE - 1 + s;
                bool ok = (k < KDIM) & (ih >= 0) & (ih < H) & (iw >= 0) & (iw < W);
                float v = ok ? (float)inb[((long)c * H + ih) * W + iw] : 0.f;
                bf[h] = (_Float16)v;
            }
        }
        // ---- MT WMMAs sharing the B fragment ----
#pragma unroll
        for (int mt = 0; mt < MT; ++mt) {
            const v16h a = *(const v16h*)(Afrag +
                (((long)ks * MTILES + (mbase_t + mt)) * 32 + lane) * 16);
            acc[mt] = __builtin_amdgcn_wmma_f32_16x16x32_f16(
                false, a, false, bf, (short)0, acc[mt], false, false);
        }
    }

    // ---- fused scale/bias/ReLU epilogue, packed NHWC f16 stores ----
    _Float16* outp = out + ((long)b * OH * OW + p) * COUT;
#pragma unroll
    for (int mt = 0; mt < MT; ++mt) {
        const int co0 = (mbase_t + mt) * 16 + 8 * g;   // D: vgpr j -> M = j + 8*g
        v8h ov;
#pragma unroll
        for (int j = 0; j < 8; ++j) {
            float y = acc[mt][j] * sc[co0 + j] + bi[co0 + j];
            ov[j] = (_Float16)fmaxf(y, 0.f);
        }
        *(v8h*)(outp + co0) = ov;
    }
}

// ---------------------------------------------------------------------------
// Global average pool over 56x56 + ternary FC 128 -> 10 (f32 out).
// h3 is NHWC: lane t reads channel t -> fully coalesced across the wave.
// ---------------------------------------------------------------------------
__global__ void pool_fc_kernel(const _Float16* __restrict__ h3,   // [B,56,56,128]
                               const _Float16* __restrict__ qfc,  // [10,128] {-1,0,1}
                               const float* __restrict__ alpha_fc,
                               const float* __restrict__ fcb,
                               float* __restrict__ outp)          // [B,10]
{
    __shared__ float pooled[128];
    const int b = blockIdx.x;
    const int t = threadIdx.x;
    const int HW = 56 * 56;
    if (t < 128) {
        const _Float16* pb = h3 + (long)b * HW * 128 + t;
        float s = 0.f;
        for (int i = 0; i < HW; ++i) s += (float)pb[(long)i * 128];
        pooled[t] = s / (float)HW;
    }
    __syncthreads();
    if (t < 10) {
        float a = alpha_fc[0];
        float s = 0.f;
        for (int c = 0; c < 128; ++c) s += pooled[c] * (float)qfc[t * 128 + c];
        outp[b * 10 + t] = a * s + fcb[t];
    }
}

// ---------------------------------------------------------------------------
extern "C" void kernel_launch(void* const* d_in, const int* in_sizes, int n_in,
                              void* d_out, int out_size, void* d_ws, size_t ws_size,
                              hipStream_t stream)
{
    const float* x    = (const float*)d_in[0];
    const float* w1   = (const float*)d_in[1];
    const float* b1   = (const float*)d_in[2];
    const float* g1   = (const float*)d_in[3];
    const float* be1  = (const float*)d_in[4];
    const float* m1   = (const float*)d_in[5];
    const float* v1   = (const float*)d_in[6];
    const float* w2   = (const float*)d_in[7];
    const float* b2   = (const float*)d_in[8];
    const float* g2   = (const float*)d_in[9];
    const float* be2  = (const float*)d_in[10];
    const float* m2   = (const float*)d_in[11];
    const float* v2   = (const float*)d_in[12];
    const float* w3   = (const float*)d_in[13];
    const float* b3   = (const float*)d_in[14];
    const float* g3   = (const float*)d_in[15];
    const float* be3  = (const float*)d_in[16];
    const float* m3   = (const float*)d_in[17];
    const float* v3   = (const float*)d_in[18];
    const float* fcw  = (const float*)d_in[19];
    const float* fcb  = (const float*)d_in[20];
    float* out = (float*)d_out;
    (void)in_sizes; (void)n_in; (void)out_size; (void)ws_size;

    char* ws = (char*)d_ws;
    size_t off = 0;
    auto carve = [&](size_t bytes) -> char* {
        char* pp = ws + off;
        off += (bytes + 255) & ~(size_t)255;
        return pp;
    };

    _Float16* h1 = (_Float16*)carve(64UL * 112 * 112 * 32 * 2);  // NHWC, reused as h3
    _Float16* h2 = (_Float16*)carve(64UL * 112 * 112 * 64 * 2);  // NHWC
    _Float16* A1 = (_Float16*)carve(1UL  * 2 * 512 * 2);
    _Float16* A2 = (_Float16*)carve(9UL  * 4 * 512 * 2);
    _Float16* A3 = (_Float16*)carve(18UL * 8 * 512 * 2);
    _Float16* qf = (_Float16*)carve(1280UL * 2);
    float* alph  = (float*)carve(4 * 4);
    float* s1 = (float*)carve(32 * 4);  float* t1 = (float*)carve(32 * 4);
    float* s2 = (float*)carve(64 * 4);  float* t2 = (float*)carve(64 * 4);
    float* s3 = (float*)carve(128 * 4); float* t3 = (float*)carve(128 * 4);
    _Float16* h3 = h1;   // conv2 consumed h1 before conv3 writes

    // --- ternarize weights (tap-major fragment swizzle + alpha) ---
    ternarize_kernel<<<1, 256, 0, stream>>>(w1,  32 * 27,  32,  3,  1,  A1, nullptr, alph + 0);
    ternarize_kernel<<<1, 256, 0, stream>>>(w2,  64 * 288, 64,  32, 9,  A2, nullptr, alph + 1);
    ternarize_kernel<<<1, 256, 0, stream>>>(w3, 128 * 576, 128, 64, 18, A3, nullptr, alph + 2);
    ternarize_kernel<<<1, 256, 0, stream>>>(fcw, 1280, 0, 0, 0, nullptr, qf, alph + 3);

    // --- fold alpha + conv bias + BN into per-channel scale/bias ---
    bnfold_kernel<<<1, 32,  0, stream>>>(alph + 0, b1, g1, be1, m1, v1, s1, t1, 32);
    bnfold_kernel<<<1, 64,  0, stream>>>(alph + 1, b2, g2, be2, m2, v2, s2, t2, 64);
    bnfold_kernel<<<1, 128, 0, stream>>>(alph + 2, b3, g3, be3, m3, v3, s3, t3, 128);

    // --- conv1: [64,3,224,224] f32 NCHW -> [64,112,112,32] f16 NHWC ---
    // pixel tiles: 64 * (112*112/16) = 50176 waves -> 6272 blocks of 8 waves
    conv_wmma_kernel<float, 3, 32, 2, 2, 224, 224, 112, 112>
        <<<dim3(6272, 1), 256, 0, stream>>>(x, A1, s1, t1, h1);

    // --- conv2: NHWC [64,112,112,32] -> [64,112,112,64] ---
    conv_wmma_kernel<_Float16, 32, 64, 4, 1, 112, 112, 112, 112>
        <<<dim3(6272, 1), 256, 0, stream>>>(h1, A2, s2, t2, h2);

    // --- conv3: NHWC [64,112,112,64] -> [64,56,56,128]; MT=8 covers Cout ---
    // pixel tiles: 64 * (56*56/16) = 12544 waves -> 1568 blocks
    conv_wmma_kernel<_Float16, 64, 128, 8, 2, 112, 112, 56, 56>
        <<<dim3(1568, 1), 256, 0, stream>>>(h2, A3, s3, t3, h3);

    // --- global avg pool + ternary FC ---
    pool_fc_kernel<<<64, 128, 0, stream>>>(h3, qf, alph + 3, fcb, out);
}
